// DLRM_5102421148471
// MI455X (gfx1250) — matrix operand order repaired and checked by reading, not compiled, not run on hardware
//
#include <hip/hip_runtime.h>
#include <hip/hip_bf16.h>

typedef __attribute__((ext_vector_type(16))) _Float16 v16h;
typedef __attribute__((ext_vector_type(8)))  _Float16 v8h;
typedef __attribute__((ext_vector_type(4)))  _Float16 v4h;
typedef __attribute__((ext_vector_type(8)))  float    v8f;

#define WMMA_F16(a, b, c) \
  __builtin_amdgcn_wmma_f32_16x16x32_f16(false, (a), false, (b), (short)0, (c), false, false)

static constexpr int BATCH   = 16384;
static constexpr int N_DENSE = 13;
static constexpr int N_SPAR  = 26;
static constexpr int EMBED   = 128;
static constexpr int NFEAT   = 27;        // 1 + 26
static constexpr int TOPK    = 512;       // 506 padded to 512
static constexpr unsigned VOCAB = 1000000u;

// ---------------------------------------------------------------------------
// Pack f32 weights [K,N] (row-major) into B-fragment order, zero-padded to Kpad.
// Block (nb,kb) of 32x16 stored as 512 halves: P[blk*512 + lane*16 + j]
//   k = kb*32 + (j/8)*16 + (lane/16)*8 + j%8,  n = nb*16 + lane%16
// ---------------------------------------------------------------------------
__global__ void pack_w_kernel(const float* __restrict__ W, _Float16* __restrict__ P,
                              int K, int N, int Kpad) {
  int tid = blockIdx.x * blockDim.x + threadIdx.x;
  int total = (Kpad >> 5) * (N >> 4) * 512;
  if (tid >= total) return;
  int j    = tid & 15;
  int lane = (tid >> 4) & 31;
  int blk  = tid >> 9;
  int kblk = Kpad >> 5;
  int nb   = blk / kblk;
  int kb   = blk - nb * kblk;
  int k = kb * 32 + (j >> 3) * 16 + (lane >> 4) * 8 + (j & 7);
  int n = nb * 16 + (lane & 15);
  float v = (k < K) ? W[(size_t)k * N + n] : 0.0f;
  P[tid] = (_Float16)v;
}

// dense cols of x -> f16 [B,32] zero padded
__global__ void prep_a0_kernel(const float* __restrict__ x, _Float16* __restrict__ A0) {
  int t = blockIdx.x * blockDim.x + threadIdx.x;
  if (t >= BATCH * 32) return;
  int b = t >> 5, k = t & 31;
  A0[t] = (k < N_DENSE) ? (_Float16)x[(size_t)b * 39 + k] : (_Float16)0.0f;
}

// categorical ids: int(x[b,13+j]) % VOCAB
__global__ void prep_ids_kernel(const float* __restrict__ x, int* __restrict__ ids) {
  int t = blockIdx.x * blockDim.x + threadIdx.x;
  if (t >= BATCH * N_SPAR) return;
  int b = t / N_SPAR, j = t - b * N_SPAR;
  unsigned v = (unsigned)(int)x[(size_t)b * 39 + N_DENSE + j];
  ids[t] = (int)(v % VOCAB);
}

// zero the pad tail topin[b, 506..511]
__global__ void zero_tail_kernel(_Float16* __restrict__ topin) {
  int t = blockIdx.x * blockDim.x + threadIdx.x;
  if (t >= BATCH * 6) return;
  int b = t / 6, j = t - b * 6;
  topin[(size_t)b * TOPK + 506 + j] = (_Float16)0.0f;
}

// ---------------------------------------------------------------------------
// Tiled WMMA GEMM: Out[M,N](f16, ld=ldo) = relu(A[M,Kpad](f16, ld=lda) * W + bias)
// One wave computes a 32x64 output tile (2 M-subtiles x 4 N-subtiles,
// 8 f32 accumulators). Weights pre-packed in B-fragment order so each lane
// issues one contiguous 32B load per (k-step, n-subtile).
// ---------------------------------------------------------------------------
__global__ __launch_bounds__(256) void gemm_wmma_kernel(
    const _Float16* __restrict__ A, int lda,
    const _Float16* __restrict__ P,
    const float* __restrict__ bias,
    _Float16* __restrict__ Out, int ldo,
    int M, int kblk, int N, int relu) {
  int wid  = (blockIdx.x * blockDim.x + threadIdx.x) >> 5;
  int lane = threadIdx.x & 31;
  int mtiles = M >> 5;                 // 32-row tiles
  int mt = wid % mtiles;
  int nt = wid / mtiles;               // 64-wide N chunk
  if (nt * 64 >= N) return;

  const _Float16* arow0 =
      A + (size_t)(mt * 32 + (lane & 15)) * lda + ((lane >> 4) * 8);
  const _Float16* arow1 = arow0 + (size_t)16 * lda;
  const _Float16* pbase = P + ((size_t)(nt * 4) * kblk << 9) + (lane << 4);
  size_t pstride = (size_t)kblk << 9;  // halves per 16-col panel

  v8f acc00 = {}, acc01 = {}, acc02 = {}, acc03 = {};
  v8f acc10 = {}, acc11 = {}, acc12 = {}, acc13 = {};
  for (int kb = 0; kb < kblk; ++kb) {
    int k0 = kb << 5;
    v8h lo0 = *(const v8h*)(arow0 + k0);
    v8h hi0 = *(const v8h*)(arow0 + k0 + 16);
    v8h lo1 = *(const v8h*)(arow1 + k0);
    v8h hi1 = *(const v8h*)(arow1 + k0 + 16);
    v16h a0 = __builtin_shufflevector(lo0, hi0, 0, 1, 2, 3, 4, 5, 6, 7,
                                      8, 9, 10, 11, 12, 13, 14, 15);
    v16h a1 = __builtin_shufflevector(lo1, hi1, 0, 1, 2, 3, 4, 5, 6, 7,
                                      8, 9, 10, 11, 12, 13, 14, 15);
    const _Float16* p = pbase + ((size_t)kb << 9);
    v16h b0 = *(const v16h*)(p);
    v16h b1 = *(const v16h*)(p + pstride);
    v16h b2 = *(const v16h*)(p + 2 * pstride);
    v16h b3 = *(const v16h*)(p + 3 * pstride);
    acc00 = WMMA_F16(a0, b0, acc00);
    acc01 = WMMA_F16(a0, b1, acc01);
    acc02 = WMMA_F16(a0, b2, acc02);
    acc03 = WMMA_F16(a0, b3, acc03);
    acc10 = WMMA_F16(a1, b0, acc10);
    acc11 = WMMA_F16(a1, b1, acc11);
    acc12 = WMMA_F16(a1, b2, acc12);
    acc13 = WMMA_F16(a1, b3, acc13);
  }

  int ncol = lane & 15;
#pragma unroll
  for (int h = 0; h < 2; ++h) {
    int rbase = mt * 32 + h * 16 + ((lane >> 4) * 8);
#pragma unroll
    for (int t = 0; t < 4; ++t) {
      v8f acc;
      if (h == 0)
        acc = (t == 0) ? acc00 : (t == 1) ? acc01 : (t == 2) ? acc02 : acc03;
      else
        acc = (t == 0) ? acc10 : (t == 1) ? acc11 : (t == 2) ? acc12 : acc13;
      int n = (nt * 4 + t) * 16 + ncol;
      if (n >= N) continue;
      float bv = bias[n];
#pragma unroll
      for (int r = 0; r < 8; ++r) {
        float v = acc[r] + bv;
        if (relu) v = v > 0.0f ? v : 0.0f;
        Out[(size_t)(rbase + r) * ldo + n] = (_Float16)v;
      }
    }
  }
}

// ---------------------------------------------------------------------------
// Fused embedding gather + dot-interaction.
// One wave per sample. Stage 32x128 f16 feature tile in LDS
// (row0 = bottom-MLP output from topin[:,0:128], rows 1..26 = gathered
//  embedding rows converted f32->f16, rows 27..31 = zero).
// Compute triu(F F^T) with 3 WMMA tiles x 4 k-steps, scatter 378 f16 values
// into topin[:,128:506].
// ---------------------------------------------------------------------------
__global__ __launch_bounds__(128) void interact_kernel(
    const float* __restrict__ emb,
    const int* __restrict__ ids,
    _Float16* __restrict__ topin) {
  __shared__ _Float16 tile[4][32][EMBED];   // 32 KB
  int wave = threadIdx.x >> 5;
  int lane = threadIdx.x & 31;
  int s = blockIdx.x * 4 + wave;            // BATCH divisible by 4

  _Float16(*T)[EMBED] = tile[wave];
  // row 0: bottom output (already f16 in topin)
  {
    const _Float16* src = topin + (size_t)s * TOPK;
    *(v4h*)&T[0][lane * 4] = *(const v4h*)(src + lane * 4);
  }
  // rows 1..26: gather + convert
  const int* sid = ids + (size_t)s * N_SPAR;
#pragma unroll 1
  for (int r = 1; r <= N_SPAR; ++r) {
    const float* er = emb + (size_t)sid[r - 1] * EMBED;
    float4 f = *(const float4*)(er + lane * 4);
    v4h h = {(_Float16)f.x, (_Float16)f.y, (_Float16)f.z, (_Float16)f.w};
    *(v4h*)&T[r][lane * 4] = h;
  }
  // rows 27..31: zero pad
  v4h z = {};
#pragma unroll
  for (int r = NFEAT; r < 32; ++r) *(v4h*)&T[r][lane * 4] = z;
  __syncthreads();

  const int mts[3] = {0, 0, 1};
  const int nts[3] = {0, 1, 1};
  _Float16* outp = topin + (size_t)s * TOPK + EMBED;

#pragma unroll
  for (int p = 0; p < 3; ++p) {
    int mt = mts[p], nt = nts[p];
    int arow = mt * 16 + (lane & 15);
    int asel = (lane >> 4) * 16;
    int brow = nt * 16 + (lane & 15);
    int bsel = (lane >> 4) * 16;
    v8f acc = {};
#pragma unroll
    for (int kb = 0; kb < 4; ++kb) {
      int k0 = kb << 5;
      v8h lo = *(const v8h*)&T[arow][k0 + asel];
      v8h hi = *(const v8h*)&T[arow][k0 + asel + 8];
      v16h a = __builtin_shufflevector(lo, hi, 0, 1, 2, 3, 4, 5, 6, 7,
                                       8, 9, 10, 11, 12, 13, 14, 15);
      v16h b = *(const v16h*)&T[brow][k0 + bsel];
      acc = WMMA_F16(a, b, acc);
    }
    int mbase = mt * 16 + ((lane >> 4) * 8);
    int n = nt * 16 + (lane & 15);
#pragma unroll
    for (int r = 0; r < 8; ++r) {
      int m = mbase + r;
      if (m < NFEAT && n < NFEAT && n >= m) {
        int idx = NFEAT * m - (m * (m - 1)) / 2 + (n - m);
        outp[idx] = (_Float16)acc[r];
      }
    }
  }
}

// final 256 -> 1 layer: per-sample dot product, wave32 shuffle reduce
__global__ __launch_bounds__(256) void final_dot_kernel(
    const _Float16* __restrict__ t3, const float* __restrict__ w,
    const float* __restrict__ b, float* __restrict__ out) {
  int wid  = (blockIdx.x * blockDim.x + threadIdx.x) >> 5;
  int lane = threadIdx.x & 31;
  if (wid >= BATCH) return;
  const _Float16* row = t3 + (size_t)wid * 256;
  float acc = 0.0f;
#pragma unroll
  for (int i = 0; i < 8; ++i) {
    int k = lane + 32 * i;
    acc += (float)row[k] * w[k];
  }
#pragma unroll
  for (int off = 16; off > 0; off >>= 1) acc += __shfl_xor(acc, off, 32);
  if (lane == 0) out[wid] = acc + b[0];
}

// ---------------------------------------------------------------------------
static inline size_t alignup(size_t v) { return (v + 255) & ~(size_t)255; }

extern "C" void kernel_launch(void* const* d_in, const int* in_sizes, int n_in,
                              void* d_out, int out_size, void* d_ws, size_t ws_size,
                              hipStream_t stream) {
  (void)in_sizes; (void)n_in; (void)out_size; (void)ws_size;
  const float* x   = (const float*)d_in[0];
  const float* bw0 = (const float*)d_in[2];
  const float* bb0 = (const float*)d_in[3];
  const float* bw1 = (const float*)d_in[4];
  const float* bb1 = (const float*)d_in[5];
  const float* bw2 = (const float*)d_in[6];
  const float* bb2 = (const float*)d_in[7];
  const float* emb = (const float*)d_in[8];
  const float* tw0 = (const float*)d_in[9];
  const float* tb0 = (const float*)d_in[10];
  const float* tw1 = (const float*)d_in[11];
  const float* tb1 = (const float*)d_in[12];
  const float* tw2 = (const float*)d_in[13];
  const float* tb2 = (const float*)d_in[14];
  const float* tw3 = (const float*)d_in[15];
  const float* tb3 = (const float*)d_in[16];
  const float* tw4 = (const float*)d_in[17];
  const float* tb4 = (const float*)d_in[18];
  float* out = (float*)d_out;

  // workspace carve-up (f16 activations + packed weights + ids)
  char* ws = (char*)d_ws;
  size_t off = 0;
  auto carve = [&](size_t bytes) { void* p = ws + off; off += alignup(bytes); return p; };
  _Float16* A0    = (_Float16*)carve((size_t)BATCH * 32 * 2);
  _Float16* act0  = (_Float16*)carve((size_t)BATCH * 512 * 2);
  _Float16* act1  = (_Float16*)carve((size_t)BATCH * 256 * 2);
  _Float16* topin = (_Float16*)carve((size_t)BATCH * TOPK * 2);
  _Float16* t0a   = (_Float16*)carve((size_t)BATCH * 1024 * 2);
  _Float16* t1a   = (_Float16*)carve((size_t)BATCH * 1024 * 2);
  _Float16* t2a   = act0;   // dead by the time top layer 2 runs (16 MB, needs 16 MB)
  _Float16* t3a   = act1;   // dead by the time top layer 3 runs (8 MB, needs 8 MB)
  int*      ids   = (int*)carve((size_t)BATCH * N_SPAR * 4);
  _Float16* pbw0  = (_Float16*)carve((size_t)32 * 512 * 2);
  _Float16* pbw1  = (_Float16*)carve((size_t)512 * 256 * 2);
  _Float16* pbw2  = (_Float16*)carve((size_t)256 * 128 * 2);
  _Float16* ptw0  = (_Float16*)carve((size_t)512 * 1024 * 2);
  _Float16* ptw1  = (_Float16*)carve((size_t)1024 * 1024 * 2);
  _Float16* ptw2  = (_Float16*)carve((size_t)1024 * 512 * 2);
  _Float16* ptw3  = (_Float16*)carve((size_t)512 * 256 * 2);

  auto pack = [&](const float* W, _Float16* P, int K, int N, int Kpad) {
    int total = (Kpad >> 5) * (N >> 4) * 512;
    pack_w_kernel<<<(total + 255) / 256, 256, 0, stream>>>(W, P, K, N, Kpad);
  };
  auto gemm = [&](const _Float16* A, int lda, const _Float16* P, const float* bias,
                  _Float16* O, int ldo, int M, int Kpad, int N, int relu) {
    int waves = (M >> 5) * (N >> 6);
    gemm_wmma_kernel<<<waves * 32 / 256, 256, 0, stream>>>(
        A, lda, P, bias, O, ldo, M, Kpad >> 5, N, relu);
  };

  // weight packing (K padded to multiple of 32)
  pack(bw0, pbw0, 13,   512,  32);
  pack(bw1, pbw1, 512,  256,  512);
  pack(bw2, pbw2, 256,  128,  256);
  pack(tw0, ptw0, 506,  1024, 512);
  pack(tw1, ptw1, 1024, 1024, 1024);
  pack(tw2, ptw2, 1024, 512,  1024);
  pack(tw3, ptw3, 512,  256,  512);

  // input prep
  prep_a0_kernel<<<(BATCH * 32 + 255) / 256, 256, 0, stream>>>(x, A0);
  prep_ids_kernel<<<(BATCH * N_SPAR + 255) / 256, 256, 0, stream>>>(x, ids);
  zero_tail_kernel<<<(BATCH * 6 + 255) / 256, 256, 0, stream>>>(topin);

  // bottom MLP: 13(->32) -> 512 -> 256 -> 128 (into topin[:,0:128])
  gemm(A0,   32,  pbw0, bb0, act0,  512,  BATCH, 32,  512, 1);
  gemm(act0, 512, pbw1, bb1, act1,  256,  BATCH, 512, 256, 1);
  gemm(act1, 256, pbw2, bb2, topin, TOPK, BATCH, 256, 128, 1);

  // fused embedding gather + dot interaction -> topin[:,128:506]
  interact_kernel<<<BATCH / 4, 128, 0, stream>>>(emb, ids, topin);

  // top MLP: 506(->512) -> 1024 -> 1024 -> 512 -> 256 -> 1
  gemm(topin, TOPK, ptw0, tb0, t0a, 1024, BATCH, 512,  1024, 1);
  gemm(t0a,   1024, ptw1, tb1, t1a, 1024, BATCH, 1024, 1024, 1);
  gemm(t1a,   1024, ptw2, tb2, t2a, 512,  BATCH, 1024, 512,  1);
  gemm(t2a,   512,  ptw3, tb3, t3a, 256,  BATCH, 512,  256,  1);
  final_dot_kernel<<<BATCH * 32 / 256, 256, 0, stream>>>(t3a, tw4, tb4, out);
}
